// xLSTMRegressor_77506979824205
// MI455X (gfx1250) — matrix operand order repaired and checked by reading, not compiled, not run on hardware
//
#include <hip/hip_runtime.h>
#include <hip/hip_bf16.h>

typedef __attribute__((ext_vector_type(16))) _Float16 v16h;
typedef __attribute__((ext_vector_type(8)))  _Float16 v8h;
typedef __attribute__((ext_vector_type(8)))  float    v8f;

#define BB  8
#define SS  512
#define FF  64
#define DD  512
#define II  1024
#define NHH 8
#define LLAYERS 2
#define KKC 4
#define DHH 128
#define MTOT 4096   // B*S

// ---------------------------------------------------------------------------
// WMMA helpers (CDNA5 gfx1250, wave32)
// ---------------------------------------------------------------------------
__device__ __forceinline__ v8f wmma32(v16h a, v16h b, v8f c) {
  return __builtin_amdgcn_wmma_f32_16x16x32_f16(false, a, false, b, (short)0, c,
                                                false, false);
}

// A fragment: 16x32 f16 tile, row-major memory with leading dim lda.
// lanes 0-15: row M=lane, halves 0..15 = K0..15
// lanes 16-31: row M=lane-16, halves 0..15 = K16..31
__device__ __forceinline__ v16h load_a_frag(const _Float16* p, int lda, int lane) {
  int row  = lane & 15;
  int koff = (lane >> 4) * 16;
  const _Float16* q = p + (size_t)row * lda;
  v8h lo = *(const v8h*)(q + koff);
  v8h hi = *(const v8h*)(q + 8 + koff);
  return __builtin_shufflevector(lo, hi, 0,1,2,3,4,5,6,7,8,9,10,11,12,13,14,15);
}

// B fragment: 32x16 f16 tile where element(kk, n) = p[n*ldk + kk]
// (i.e. memory holds the N x K transposed matrix -> contiguous per-lane loads)
// lanes 0-15: col N=lane, K 0..15 ; lanes 16-31: col N=lane-16, K 16..31
__device__ __forceinline__ v16h load_bT_frag(const _Float16* p, int ldk, int lane) {
  int col  = lane & 15;
  int krow = (lane >> 4) * 16;
  return *(const v16h*)(p + (size_t)col * ldk + krow);
}

// ---------------------------------------------------------------------------
// Generic WMMA GEMM: C[M,N] = A[M,K] * W[K,N] ; W stored transposed (N x K) f16
// block = 128 threads (4 waves); wave w handles rows [m0,m0+16), cols [n0,n0+64)
// mode 0: outF = acc + bias[col]        (f32)
// mode 1: outH = (f16)acc
// mode 2: outF += acc                   (residual accumulate)
// mode 3: outH transposed per-head:  vT[((row/512)*1024 + col)*512 + row%512]
// ---------------------------------------------------------------------------
__global__ __launch_bounds__(128) void gemm_wmma(
    const _Float16* __restrict__ A, int lda,
    const _Float16* __restrict__ WT, int Kdim,
    const float* __restrict__ bias,
    float* __restrict__ outF, _Float16* __restrict__ outH,
    int ldo, int mode)
{
  int lane = threadIdx.x & 31;
  int wave = threadIdx.x >> 5;
  int m0 = blockIdx.y * 64 + wave * 16;
  int n0 = blockIdx.x * 64;

  v8f acc[4];
#pragma unroll
  for (int j = 0; j < 4; ++j) { v8f z = {}; acc[j] = z; }

  for (int k0 = 0; k0 < Kdim; k0 += 32) {
    v16h a = load_a_frag(A + (size_t)m0 * lda + k0, lda, lane);
#pragma unroll
    for (int j = 0; j < 4; ++j) {
      v16h b = load_bT_frag(WT + (size_t)(n0 + j * 16) * Kdim + k0, Kdim, lane);
      acc[j] = wmma32(a, b, acc[j]);
    }
  }

  int col = lane & 15, hlf = lane >> 4;
#pragma unroll
  for (int j = 0; j < 4; ++j) {
    int cg = n0 + j * 16 + col;
#pragma unroll
    for (int r = 0; r < 8; ++r) {
      int rg = m0 + r + hlf * 8;
      float v = acc[j][r];
      if (mode == 0)      outF[(size_t)rg * ldo + cg] = v + bias[cg];
      else if (mode == 1) outH[(size_t)rg * ldo + cg] = (_Float16)v;
      else if (mode == 2) outF[(size_t)rg * ldo + cg] += v;
      else {
        int bb = rg >> 9, s = rg & 511;
        outH[((size_t)bb * II + cg) * SS + s] = (_Float16)v;
      }
    }
  }
}

// ---------------------------------------------------------------------------
// Weight convert + transpose: dst[n*K+k] = (f16) src[k*N+n]
// ---------------------------------------------------------------------------
__global__ __launch_bounds__(256) void convT_kernel(const float* __restrict__ src,
                                                    _Float16* __restrict__ dst,
                                                    int Kdim, int N) {
  int idx = blockIdx.x * 256 + threadIdx.x;
  if (idx >= N * Kdim) return;
  int n = idx / Kdim, k = idx - n * Kdim;
  dst[idx] = (_Float16)src[(size_t)k * N + n];
}

__global__ __launch_bounds__(256) void tof16_kernel(const float* __restrict__ src,
                                                    _Float16* __restrict__ dst, int n) {
  int idx = blockIdx.x * 256 + threadIdx.x;
  if (idx < n) dst[idx] = (_Float16)src[idx];
}

// ---------------------------------------------------------------------------
// Row layernorm over D=512 -> f16 output. grid = 4096, block = 256
// ---------------------------------------------------------------------------
__global__ __launch_bounds__(256) void ln_f16_kernel(const float* __restrict__ x,
    const float* __restrict__ w, const float* __restrict__ b,
    _Float16* __restrict__ out)
{
  int row = blockIdx.x, t = threadIdx.x;
  const float* xr = x + (size_t)row * DD;
  float x0 = xr[t], x1 = xr[t + 256];
  __shared__ float red[256];
  red[t] = x0 + x1; __syncthreads();
  for (int off = 128; off > 0; off >>= 1) { if (t < off) red[t] += red[t + off]; __syncthreads(); }
  float mu = red[0] * (1.f / DD);
  __syncthreads();
  float d0 = x0 - mu, d1 = x1 - mu;
  red[t] = d0 * d0 + d1 * d1; __syncthreads();
  for (int off = 128; off > 0; off >>= 1) { if (t < off) red[t] += red[t + off]; __syncthreads(); }
  float rstd = rsqrtf(red[0] * (1.f / DD) + 1e-5f);
  _Float16* o = out + (size_t)row * DD;
  o[t]       = (_Float16)(d0 * rstd * w[t] + b[t]);
  o[t + 256] = (_Float16)(d1 * rstd * w[t + 256] + b[t + 256]);
}

// ---------------------------------------------------------------------------
// Causal depthwise conv (K=4) + SiLU.  x_m lives in up16 (row stride 2048).
// ---------------------------------------------------------------------------
__global__ __launch_bounds__(256) void conv_kernel(const _Float16* __restrict__ xm,
    const float* __restrict__ cw, const float* __restrict__ cb,
    _Float16* __restrict__ ca16)
{
  int idx = blockIdx.x * 256 + threadIdx.x;     // < 4096*1024
  int i  = idx & (II - 1);
  int bs = idx >> 10;
  int s  = bs & (SS - 1);
  float acc = cb[i];
#pragma unroll
  for (int j = 0; j < KKC; ++j) {
    int so = s + j - (KKC - 1);
    if (so >= 0)
      acc += (float)xm[(size_t)(bs + j - (KKC - 1)) * 2048 + i] * cw[i * KKC + j];
  }
  float sv = acc / (1.f + __expf(-acc));
  ca16[idx] = (_Float16)sv;
}

// ---------------------------------------------------------------------------
// Gate GEMV: ig/fg[b,h,s] = qkv(b,s,:) . w[:,h] + bias[h]
// grid = 4096 (one per b,s); block = 512 (16 waves: 8 ig heads, 8 fg heads)
// ---------------------------------------------------------------------------
__global__ __launch_bounds__(512) void gates_kernel(const _Float16* __restrict__ q16,
    const _Float16* __restrict__ k16, const _Float16* __restrict__ vT16,
    const float* __restrict__ igw, const float* __restrict__ igb,
    const float* __restrict__ fgw, const float* __restrict__ fgb,
    float* __restrict__ ig, float* __restrict__ fg)
{
  int bs = blockIdx.x; int b = bs >> 9, s = bs & 511;
  int lane = threadIdx.x & 31, wave = threadIdx.x >> 5;
  int head = wave & 7;
  bool isFg = wave >= 8;
  const float* w = isFg ? fgw : igw;
  float acc = 0.f;
  for (int e = lane; e < 3 * II; e += 32) {
    float qv;
    if (e < II)            qv = (float)q16[(size_t)bs * II + e];
    else if (e < 2 * II)   qv = (float)k16[(size_t)bs * II + (e - II)];
    else {
      int e2 = e - 2 * II;
      qv = (float)vT16[(((size_t)b * NHH + (e2 >> 7)) * DHH + (e2 & 127)) * SS + s];
    }
    acc += qv * w[(size_t)e * NHH + head];
  }
#pragma unroll
  for (int m = 16; m >= 1; m >>= 1) acc += __shfl_xor(acc, m, 32);
  if (lane == 0) {
    float* o = isFg ? fg : ig;
    float bv = (isFg ? fgb : igb)[head];
    o[((size_t)b * NHH + head) * SS + s] = acc + bv;
  }
}

// ---------------------------------------------------------------------------
// Per-(b,h): cs = cumsum(log_sigmoid(fg));  maxd[s] = cs[s]+max_{t<=s}(ig[t]-cs[t])
// grid = 64, block = 512
// ---------------------------------------------------------------------------
__global__ __launch_bounds__(512) void scan_kernel(const float* __restrict__ ig,
    const float* __restrict__ fg, float* __restrict__ cs, float* __restrict__ maxd)
{
  int bh = blockIdx.x; int t = threadIdx.x;
  __shared__ float buf[512];
  float f = fg[(size_t)bh * SS + t];
  float ls = (f < 0.f) ? (f - log1pf(__expf(f))) : (-log1pf(__expf(-f)));
  buf[t] = ls; __syncthreads();
  for (int off = 1; off < 512; off <<= 1) {
    float v = buf[t];
    float u = (t >= off) ? buf[t - off] : 0.f;
    __syncthreads();
    buf[t] = v + u;
    __syncthreads();
  }
  float c = buf[t];
  cs[(size_t)bh * SS + t] = c;
  __syncthreads();
  buf[t] = ig[(size_t)bh * SS + t] - c;
  __syncthreads();
  for (int off = 1; off < 512; off <<= 1) {
    float v = buf[t];
    float u = (t >= off) ? buf[t - off] : -3.4e38f;
    __syncthreads();
    buf[t] = fmaxf(v, u);
    __syncthreads();
  }
  maxd[(size_t)bh * SS + t] = c + buf[t];
}

// ---------------------------------------------------------------------------
// mLSTM attention, one (b,h) per blockIdx.x, 64 query rows per blockIdx.y.
// 4 independent waves; each wave owns a 16-row strip and its own LDS P-tile.
// ---------------------------------------------------------------------------
__global__ __launch_bounds__(128) void attn_kernel(const _Float16* __restrict__ q16,
    const _Float16* __restrict__ k16, const _Float16* __restrict__ vT16,
    const float* __restrict__ cs, const float* __restrict__ ig,
    const float* __restrict__ maxd, float* __restrict__ hc)
{
  int bh = blockIdx.x; int b = bh >> 3, h = bh & 7;
  int lane = threadIdx.x & 31, wave = threadIdx.x >> 5;
  int r0 = blockIdx.y * 64 + wave * 16;
  const _Float16* qh = q16 + (size_t)b * SS * II + h * DHH;
  const _Float16* kh = k16 + (size_t)b * SS * II + h * DHH;
  const _Float16* vh = vT16 + (size_t)bh * DHH * SS;   // [DH][S]
  const float* csb = cs   + (size_t)bh * SS;
  const float* igb = ig   + (size_t)bh * SS;
  const float* mdb = maxd + (size_t)bh * SS;

  __shared__ _Float16 pbuf[4][16 * 32];
  _Float16* pt = pbuf[wave];

  int col = lane & 15, hlf = lane >> 4;
  v8f out[8];
#pragma unroll
  for (int n = 0; n < 8; ++n) { v8f z = {}; out[n] = z; }
  float rsum[8], mycs[8], mymd[8];
#pragma unroll
  for (int r = 0; r < 8; ++r) {
    rsum[r] = 0.f;
    int srow = r0 + r + hlf * 8;
    mycs[r] = csb[srow];
    mymd[r] = mdb[srow];
  }
  const float scale = 0.08838834764831845f;   // 1/sqrt(128)

  for (int t0 = 0; t0 < r0 + 16; t0 += 32) {
    v8f s0 = {}; v8f s1 = {};
#pragma unroll
    for (int kd = 0; kd < DHH; kd += 32) {
      v16h a  = load_a_frag(qh + (size_t)r0 * II + kd, II, lane);
      v16h b0 = load_bT_frag(kh + (size_t)t0 * II + kd, II, lane);
      s0 = wmma32(a, b0, s0);
      v16h b1 = load_bT_frag(kh + (size_t)(t0 + 16) * II + kd, II, lane);
      s1 = wmma32(a, b1, s1);
    }
    float cst0 = csb[t0 + col],      igt0 = igb[t0 + col];
    float cst1 = csb[t0 + 16 + col], igt1 = igb[t0 + 16 + col];
#pragma unroll
    for (int r = 0; r < 8; ++r) {
      int srow = r0 + r + hlf * 8;
      float p0 = (t0 + col <= srow)
                   ? s0[r] * scale * __expf(mycs[r] - cst0 + igt0 - mymd[r]) : 0.f;
      float p1 = (t0 + 16 + col <= srow)
                   ? s1[r] * scale * __expf(mycs[r] - cst1 + igt1 - mymd[r]) : 0.f;
      rsum[r] += p0 + p1;
      pt[(r + hlf * 8) * 32 + col]      = (_Float16)p0;
      pt[(r + hlf * 8) * 32 + 16 + col] = (_Float16)p1;
    }
    asm volatile("s_wait_dscnt 0" ::: "memory");   // P stores visible to frag load
    v16h pa = load_a_frag(pt, 32, lane);
#pragma unroll
    for (int n = 0; n < 8; ++n) {
      v16h vb = load_bT_frag(vh + (size_t)(n * 16) * SS + t0, SS, lane);
      out[n] = wmma32(pa, vb, out[n]);
    }
  }

#pragma unroll
  for (int r = 0; r < 8; ++r) {   // reduce row-sums across the 16 lanes of each half
    float v = rsum[r];
    v += __shfl_xor(v, 1, 32); v += __shfl_xor(v, 2, 32);
    v += __shfl_xor(v, 4, 32); v += __shfl_xor(v, 8, 32);
    rsum[r] = v;
  }
  float* hcp = hc + (size_t)b * SS * II + h * DHH;
#pragma unroll
  for (int r = 0; r < 8; ++r) {
    int srow = r0 + r + hlf * 8;
    float denom = fmaxf(fabsf(rsum[r]), __expf(-mymd[r])) + 1e-6f;
    float inv = 1.f / denom;
#pragma unroll
    for (int n = 0; n < 8; ++n)
      hcp[(size_t)srow * II + n * 16 + col] = out[n][r] * inv;
  }
}

// ---------------------------------------------------------------------------
// Group-norm over DH per head + skip + silu(z) gate -> hs16
// grid = 4096 (one per b,s); block = 256 (wave per head)
// ---------------------------------------------------------------------------
__global__ __launch_bounds__(256) void combine_kernel(const float* __restrict__ hc,
    const _Float16* __restrict__ ca16, const _Float16* __restrict__ up16,
    const float* __restrict__ gnw, const float* __restrict__ skw,
    _Float16* __restrict__ hs16)
{
  int bs = blockIdx.x;
  int lane = threadIdx.x & 31, head = threadIdx.x >> 5;
  const float* hr = hc + (size_t)bs * II + head * DHH;
  float x[4], sum = 0.f, sq = 0.f;
#pragma unroll
  for (int j = 0; j < 4; ++j) { x[j] = hr[lane + 32 * j]; sum += x[j]; sq += x[j] * x[j]; }
#pragma unroll
  for (int m = 16; m >= 1; m >>= 1) { sum += __shfl_xor(sum, m, 32); sq += __shfl_xor(sq, m, 32); }
  float mu = sum * (1.f / DHH);
  float var = sq * (1.f / DHH) - mu * mu;
  float rstd = rsqrtf(var + 1e-5f);
#pragma unroll
  for (int j = 0; j < 4; ++j) {
    int i = head * DHH + lane + 32 * j;
    float hn = (x[j] - mu) * rstd * gnw[i];
    float cav = (float)ca16[(size_t)bs * II + i];
    float zz = (float)up16[(size_t)bs * 2048 + II + i];
    float sz = zz / (1.f + __expf(-zz));
    hs16[(size_t)bs * II + i] = (_Float16)((hn + skw[i] * cav) * sz);
  }
}

// ---------------------------------------------------------------------------
// Final: post-LN on last token + 2-layer MLP head.  grid = 8, block = 256
// ---------------------------------------------------------------------------
__global__ __launch_bounds__(256) void head_kernel(const float* __restrict__ h,
    const float* __restrict__ pw, const float* __restrict__ pb,
    const float* __restrict__ h1w, const float* __restrict__ h1b,
    const float* __restrict__ h2w, const float* __restrict__ h2b,
    float* __restrict__ out)
{
  int b = blockIdx.x, t = threadIdx.x;
  const float* row = h + ((size_t)b * SS + (SS - 1)) * DD;
  __shared__ float red[256];
  __shared__ float nrm[512];
  __shared__ float hid[32];
  float x0 = row[t], x1 = row[t + 256];
  red[t] = x0 + x1; __syncthreads();
  for (int off = 128; off > 0; off >>= 1) { if (t < off) red[t] += red[t + off]; __syncthreads(); }
  float mu = red[0] * (1.f / DD);
  __syncthreads();
  float d0 = x0 - mu, d1 = x1 - mu;
  red[t] = d0 * d0 + d1 * d1; __syncthreads();
  for (int off = 128; off > 0; off >>= 1) { if (t < off) red[t] += red[t + off]; __syncthreads(); }
  float rstd = rsqrtf(red[0] * (1.f / DD) + 1e-5f);
  __syncthreads();
  nrm[t]       = d0 * rstd * pw[t] + pb[t];
  nrm[t + 256] = d1 * rstd * pw[t + 256] + pb[t + 256];
  __syncthreads();
  if (t < 32) {
    float a = h1b[t];
    for (int e = 0; e < 512; ++e) a += nrm[e] * h1w[e * 32 + t];
    hid[t] = fmaxf(a, 0.f);
  }
  __syncthreads();
  if (t < 24) {
    float a = h2b[t];
    for (int e = 0; e < 32; ++e) a += hid[e] * h2w[e * 24 + t];
    out[b * 24 + t] = a;
  }
}

// ---------------------------------------------------------------------------
extern "C" void kernel_launch(void* const* d_in, const int* in_sizes, int n_in,
                              void* d_out, int out_size, void* d_ws, size_t ws_size,
                              hipStream_t stream)
{
  (void)in_sizes; (void)n_in; (void)out_size;
  const float* x    = (const float*)d_in[0];
  const float* in_w = (const float*)d_in[1];
  const float* in_b = (const float*)d_in[2];
  const float* ln_w = (const float*)d_in[3];
  const float* ln_b = (const float*)d_in[4];
  const float* up_w = (const float*)d_in[5];
  const float* cv_w = (const float*)d_in[6];
  const float* cv_b = (const float*)d_in[7];
  const float* q_w  = (const float*)d_in[8];
  const float* k_w  = (const float*)d_in[9];
  const float* v_w  = (const float*)d_in[10];
  const float* ig_w = (const float*)d_in[11];
  const float* ig_b = (const float*)d_in[12];
  const float* fg_w = (const float*)d_in[13];
  const float* fg_b = (const float*)d_in[14];
  const float* gn_w = (const float*)d_in[15];
  const float* sk_w = (const float*)d_in[16];
  const float* dn_w = (const float*)d_in[17];
  const float* po_w = (const float*)d_in[18];
  const float* po_b = (const float*)d_in[19];
  const float* h1w  = (const float*)d_in[20];
  const float* h1b  = (const float*)d_in[21];
  const float* h2w  = (const float*)d_in[22];
  const float* h2b  = (const float*)d_in[23];
  float* out = (float*)d_out;

  char* ws = (char*)d_ws;
  size_t off = 0;
  auto alloc = [&](size_t bytes) -> char* {
    char* p = ws + off;
    off = (off + bytes + 255) & ~(size_t)255;
    return p;
  };
  _Float16* WT_in = (_Float16*)alloc((size_t)DD * FF * 2);
  _Float16* WT_up = (_Float16*)alloc((size_t)LLAYERS * 2 * II * DD * 2);
  _Float16* WT_q  = (_Float16*)alloc((size_t)LLAYERS * II * II * 2);
  _Float16* WT_k  = (_Float16*)alloc((size_t)LLAYERS * II * II * 2);
  _Float16* WT_v  = (_Float16*)alloc((size_t)LLAYERS * II * II * 2);
  _Float16* WT_dn = (_Float16*)alloc((size_t)LLAYERS * DD * II * 2);
  _Float16* x16   = (_Float16*)alloc((size_t)MTOT * FF * 2);
  float*    hbuf  = (float*)   alloc((size_t)MTOT * DD * 4);
  _Float16* xn16  = (_Float16*)alloc((size_t)MTOT * DD * 2);
  _Float16* up16  = (_Float16*)alloc((size_t)MTOT * 2 * II * 2);
  _Float16* ca16  = (_Float16*)alloc((size_t)MTOT * II * 2);
  _Float16* q16b  = (_Float16*)alloc((size_t)MTOT * II * 2);
  _Float16* k16b  = (_Float16*)alloc((size_t)MTOT * II * 2);
  _Float16* vT16b = (_Float16*)alloc((size_t)MTOT * II * 2);
  float*    igbuf = (float*)   alloc((size_t)BB * NHH * SS * 4);
  float*    fgbuf = (float*)   alloc((size_t)BB * NHH * SS * 4);
  float*    csbuf = (float*)   alloc((size_t)BB * NHH * SS * 4);
  float*    mdbuf = (float*)   alloc((size_t)BB * NHH * SS * 4);
  float*    hcbuf = (float*)   alloc((size_t)MTOT * II * 4);
  _Float16* hs16  = (_Float16*)alloc((size_t)MTOT * II * 2);
  if (off > ws_size) return;   // workspace too small: bail deterministically

  auto cdiv = [](int a, int b) { return (a + b - 1) / b; };

  // ---- weight conversion (f32 -> f16, N x K transposed) ----
  convT_kernel<<<cdiv(DD * FF, 256), 256, 0, stream>>>(in_w, WT_in, FF, DD);
  for (int l = 0; l < LLAYERS; ++l) {
    convT_kernel<<<cdiv(2 * II * DD, 256), 256, 0, stream>>>(
        up_w + (size_t)l * DD * 2 * II, WT_up + (size_t)l * 2 * II * DD, DD, 2 * II);
    convT_kernel<<<cdiv(II * II, 256), 256, 0, stream>>>(
        q_w + (size_t)l * II * II, WT_q + (size_t)l * II * II, II, II);
    convT_kernel<<<cdiv(II * II, 256), 256, 0, stream>>>(
        k_w + (size_t)l * II * II, WT_k + (size_t)l * II * II, II, II);
    convT_kernel<<<cdiv(II * II, 256), 256, 0, stream>>>(
        v_w + (size_t)l * II * II, WT_v + (size_t)l * II * II, II, II);
    convT_kernel<<<cdiv(DD * II, 256), 256, 0, stream>>>(
        dn_w + (size_t)l * II * DD, WT_dn + (size_t)l * DD * II, II, DD);
  }
  tof16_kernel<<<cdiv(MTOT * FF, 256), 256, 0, stream>>>(x, x16, MTOT * FF);

  // ---- h = x @ in_w + in_b ----
  gemm_wmma<<<dim3(DD / 64, MTOT / 64), 128, 0, stream>>>(
      x16, FF, WT_in, FF, in_b, hbuf, nullptr, DD, 0);

  for (int l = 0; l < LLAYERS; ++l) {
    ln_f16_kernel<<<MTOT, 256, 0, stream>>>(hbuf, ln_w + l * DD, ln_b + l * DD, xn16);
    gemm_wmma<<<dim3(2 * II / 64, MTOT / 64), 128, 0, stream>>>(
        xn16, DD, WT_up + (size_t)l * 2 * II * DD, DD, nullptr, nullptr, up16, 2 * II, 1);
    conv_kernel<<<cdiv(MTOT * II, 256), 256, 0, stream>>>(
        up16, cv_w + (size_t)l * II * KKC, cv_b + l * II, ca16);
    gemm_wmma<<<dim3(II / 64, MTOT / 64), 128, 0, stream>>>(
        ca16, II, WT_q + (size_t)l * II * II, II, nullptr, nullptr, q16b, II, 1);
    gemm_wmma<<<dim3(II / 64, MTOT / 64), 128, 0, stream>>>(
        ca16, II, WT_k + (size_t)l * II * II, II, nullptr, nullptr, k16b, II, 1);
    gemm_wmma<<<dim3(II / 64, MTOT / 64), 128, 0, stream>>>(
        up16, 2 * II, WT_v + (size_t)l * II * II, II, nullptr, nullptr, vT16b, II, 3);
    gates_kernel<<<MTOT, 512, 0, stream>>>(
        q16b, k16b, vT16b,
        ig_w + (size_t)l * 3 * II * NHH, ig_b + l * NHH,
        fg_w + (size_t)l * 3 * II * NHH, fg_b + l * NHH, igbuf, fgbuf);
    scan_kernel<<<BB * NHH, 512, 0, stream>>>(igbuf, fgbuf, csbuf, mdbuf);
    attn_kernel<<<dim3(BB * NHH, SS / 64), 128, 0, stream>>>(
        q16b, k16b, vT16b, csbuf, igbuf, mdbuf, hcbuf);
    combine_kernel<<<MTOT, 256, 0, stream>>>(
        hcbuf, ca16, up16, gn_w + l * II, sk_w + l * II, hs16);
    gemm_wmma<<<dim3(DD / 64, MTOT / 64), 128, 0, stream>>>(
        hs16, II, WT_dn + (size_t)l * DD * II, II, nullptr, hbuf, nullptr, DD, 2);
  }

  head_kernel<<<BB, 256, 0, stream>>>(hbuf, po_w, po_b, h1w, h1b, h2w, h2b, out);
}
